// ExpertParallelMoE_48369921687707
// MI455X (gfx1250) — matrix operand order repaired and checked by reading, not compiled, not run on hardware
//
#include <hip/hip_runtime.h>
#include <hip/hip_bf16.h>

#define D_MODEL   1024
#define D_FF      4096
#define N_EXPERTS 8
#define N_TOKENS  4096

typedef __attribute__((ext_vector_type(16))) __bf16 v16bf;
typedef __attribute__((ext_vector_type(8)))  __bf16 v8bf;
typedef __attribute__((ext_vector_type(2)))  __bf16 v2bf;
typedef __attribute__((ext_vector_type(8)))  float  v8f;

// scalar float -> bf16 (round-half-up; cheap: one add + shift)
__device__ __forceinline__ unsigned short f2bf(float f) {
  return (unsigned short)((__float_as_uint(f) + 0x8000u) >> 16);
}

// pack two floats -> packed bf16 pair (lo = a, hi = b)
__device__ __forceinline__ unsigned int cvt2bf(float a, float b) {
#if __has_builtin(__builtin_amdgcn_cvt_pk_bf16_f32)
  v2bf r = __builtin_amdgcn_cvt_pk_bf16_f32(a, b);
  return __builtin_bit_cast(unsigned int, r);
#else
  // round-half-up both, then one v_perm_b32 picks the two high halves:
  // dst[15:0] = ra[31:16], dst[31:16] = rb[31:16]
  unsigned int ra = __float_as_uint(a) + 0x8000u;
  unsigned int rb = __float_as_uint(b) + 0x8000u;
  return __builtin_amdgcn_perm(rb, ra, 0x07060302u);
#endif
}

__device__ __forceinline__ void wait_async0() {
#if __has_builtin(__builtin_amdgcn_s_wait_asynccnt)
  __builtin_amdgcn_s_wait_asynccnt(0);
#else
  asm volatile("s_wait_asynccnt 0x0" ::: "memory");
#endif
}

// ---------------- gating / routing ----------------

__global__ void moe_init_kernel(int* counts) {
  if (threadIdx.x < N_EXPERTS) counts[threadIdx.x] = 0;
}

__global__ void moe_gate_kernel(const float* __restrict__ x,
                                const float* __restrict__ Wg,
                                const float* __restrict__ bg,
                                int* __restrict__ eArr,
                                int* __restrict__ counts) {
  int t = blockIdx.x * blockDim.x + threadIdx.x;
  if (t >= N_TOKENS) return;
  float acc[N_EXPERTS];
#pragma unroll
  for (int j = 0; j < N_EXPERTS; ++j) acc[j] = bg[j];
  const float* xr = x + (size_t)t * D_MODEL;
  for (int d = 0; d < D_MODEL; ++d) {
    float xv = xr[d];
#pragma unroll
    for (int j = 0; j < N_EXPERTS; ++j) acc[j] += xv * Wg[d * N_EXPERTS + j];
  }
  int best = 0;
  float bv = acc[0];
#pragma unroll
  for (int j = 1; j < N_EXPERTS; ++j)
    if (acc[j] > bv) { bv = acc[j]; best = j; }   // first-max tie-break = jnp.argmax
  eArr[t] = best;
  atomicAdd(&counts[best], 1);
}

__global__ void moe_offsets_kernel(const int* __restrict__ counts,
                                   int* __restrict__ offs) {
  if (threadIdx.x == 0) {
    int s = 0;
    for (int j = 0; j < N_EXPERTS; ++j) { offs[j] = s; s += counts[j]; }
  }
}

// stable grouped order: perm[pos] = token index feeding output row pos
__global__ void moe_perm_kernel(const int* __restrict__ eArr,
                                const int* __restrict__ offs,
                                int* __restrict__ perm) {
  int t = blockIdx.x * blockDim.x + threadIdx.x;
  if (t >= N_TOKENS) return;
  int my = eArr[t];
  int rank = 0;
  for (int u = 0; u < t; ++u) rank += (eArr[u] == my) ? 1 : 0;
  perm[offs[my] + rank] = t;
}

// ---------------- grouped GEMM (bf16 WMMA, f32 accumulate) ----------------
// PASS 1: H = relu(x[perm] * W1[e] + b1[e])   (A fp32 gathered, out bf16)
// PASS 2: out = H * W2[e] + b2[e]             (A bf16 contiguous via async-LDS, out fp32)

template <int PASS>
__global__ __launch_bounds__(256)
void moe_ffn_kernel(const float* __restrict__ xf,
                    const unsigned short* __restrict__ Hin,
                    const float* __restrict__ W,
                    const float* __restrict__ bias,
                    const int* __restrict__ perm,
                    const int* __restrict__ counts,
                    const int* __restrict__ offs,
                    unsigned short* __restrict__ Hout,
                    float* __restrict__ out) {
  constexpr int KDIM = (PASS == 1) ? D_MODEL : D_FF;
  constexpr int NCOL = (PASS == 1) ? D_FF : D_MODEL;

  const int eId = blockIdx.z;
  const int cnt = counts[eId];
  const int rowBase = blockIdx.y * 64;
  if (rowBase >= cnt) return;                      // block-uniform early exit
  int rowsHere = cnt - rowBase;
  if (rowsHere > 64) rowsHere = 64;
  const int posBase = offs[eId] + rowBase;
  const int colBase = blockIdx.x * 128;
  const float* Wm = W + (size_t)eId * KDIM * NCOL;

  // ping-pong LDS: A tile 64x32 bf16, B tile transposed [col][k] with padded stride
  __shared__ __align__(16) unsigned short As[2][64 * 32];
  __shared__ __align__(16) unsigned short Bs[2][128 * 40];

  const int tid  = threadIdx.x;
  const int lane = tid & 31;
  const int wave = tid >> 5;
  const int wm   = wave & 3;        // 4 row strips of 16
  const int wn   = wave >> 2;       // 2 column halves of 64
  const int lm   = lane & 15;
  const int half = lane >> 4;

  // staging coords (loop-invariant)
  const int ar  = tid >> 2;          // 0..63  A row
  const int ak  = (tid & 3) * 8;     // A k offset within 32
  const int bkp = tid >> 4;          // 0..15  B k-pair (k = 2*bkp, 2*bkp+1)
  const int bc8 = (tid & 15) * 8;    // B col offset within 128 (8 cols/thread)

  // loop-invariant source bases
  int aTok = -1;
  if (PASS == 1 && ar < rowsHere) aTok = perm[posBase + ar];
  const float* aSrc1 =
      (PASS == 1 && aTok >= 0) ? (xf + (size_t)aTok * D_MODEL + ak) : nullptr;
  const unsigned short* aSrc2 =
      (PASS == 2) ? (Hin + (size_t)(posBase + ar) * D_FF + ak) : nullptr;
  const float* bSrc = Wm + (size_t)(2 * bkp) * NCOL + colBase + bc8;

  auto stage = [&](int k0, int buf) {
    // ---- B tile: rows (k0+2*bkp, +1), 8 cols; pack pairs along K, b32 LDS stores
    {
      const float* r0 = bSrc + (size_t)k0 * NCOL;
      const float* r1 = r0 + NCOL;
      float4 a0 = ((const float4*)r0)[0];
      float4 a1 = ((const float4*)r0)[1];
      float4 c0 = ((const float4*)r1)[0];
      float4 c1 = ((const float4*)r1)[1];
      unsigned short* bs = &Bs[buf][0];
      const int kk = 2 * bkp;
      *(unsigned int*)&bs[(bc8 + 0) * 40 + kk] = cvt2bf(a0.x, c0.x);
      *(unsigned int*)&bs[(bc8 + 1) * 40 + kk] = cvt2bf(a0.y, c0.y);
      *(unsigned int*)&bs[(bc8 + 2) * 40 + kk] = cvt2bf(a0.z, c0.z);
      *(unsigned int*)&bs[(bc8 + 3) * 40 + kk] = cvt2bf(a0.w, c0.w);
      *(unsigned int*)&bs[(bc8 + 4) * 40 + kk] = cvt2bf(a1.x, c1.x);
      *(unsigned int*)&bs[(bc8 + 5) * 40 + kk] = cvt2bf(a1.y, c1.y);
      *(unsigned int*)&bs[(bc8 + 6) * 40 + kk] = cvt2bf(a1.z, c1.z);
      *(unsigned int*)&bs[(bc8 + 7) * 40 + kk] = cvt2bf(a1.w, c1.w);
      if (k0 + 64 < KDIM)
        __builtin_prefetch(bSrc + (size_t)(k0 + 64) * NCOL, 0, 0);
    }
    // ---- A tile
    if constexpr (PASS == 1) {
      unsigned int p0, p1, p2, p3;
      if (aTok >= 0) {
        const float4* s = (const float4*)(aSrc1 + k0);
        float4 v0 = s[0], v1 = s[1];
        p0 = cvt2bf(v0.x, v0.y); p1 = cvt2bf(v0.z, v0.w);
        p2 = cvt2bf(v1.x, v1.y); p3 = cvt2bf(v1.z, v1.w);
      } else {
        p0 = p1 = p2 = p3 = 0u;
      }
      *(uint4*)&As[buf][ar * 32 + ak] = make_uint4(p0, p1, p2, p3);
    } else {
      if (ar < rowsHere) {
        // already bf16 in memory: async DMA straight into LDS (ASYNCcnt path)
        unsigned lds = (unsigned)(uintptr_t)&As[buf][ar * 32 + ak];
        unsigned long long g = (unsigned long long)(uintptr_t)(aSrc2 + k0);
        asm volatile("global_load_async_to_lds_b128 %0, %1, off"
                     :: "v"(lds), "v"(g) : "memory");
      } else {
        *(uint4*)&As[buf][ar * 32 + ak] = make_uint4(0u, 0u, 0u, 0u);
      }
    }
  };

  v8f acc[4];
#pragma unroll
  for (int t = 0; t < 4; ++t) {
    v8f z = {0.f, 0.f, 0.f, 0.f, 0.f, 0.f, 0.f, 0.f};
    acc[t] = z;
  }

  const int amRow = wm * 16 + lm;
  const int nBase = wn * 64 + lm;

  stage(0, 0);
  int buf = 0;
  for (int k0 = 0; k0 < KDIM; k0 += 32) {
    if constexpr (PASS == 2) wait_async0();   // our async LDS writes landed
    __syncthreads();                          // staged buffer visible block-wide
    if (k0 + 32 < KDIM) stage(k0 + 32, buf ^ 1);  // overlap next stage w/ compute

    // ---- batched fragment loads, then 4 WMMAs
    const unsigned short* as = &As[buf][0];
    const unsigned short* bs = &Bs[buf][0];
    // A layout: lane(m=lm, half): elem j -> k = 16*half + j
    v8bf aLo = *(const v8bf*)&as[amRow * 32 + 16 * half];
    v8bf aHi = *(const v8bf*)&as[amRow * 32 + 16 * half + 8];
    // B layout: lane(n, half): elem j -> k = 16*half + j
    v8bf b0l = *(const v8bf*)&bs[(nBase +  0) * 40 + 16 * half];
    v8bf b0h = *(const v8bf*)&bs[(nBase +  0) * 40 + 16 * half + 8];
    v8bf b1l = *(const v8bf*)&bs[(nBase + 16) * 40 + 16 * half];
    v8bf b1h = *(const v8bf*)&bs[(nBase + 16) * 40 + 16 * half + 8];
    v8bf b2l = *(const v8bf*)&bs[(nBase + 32) * 40 + 16 * half];
    v8bf b2h = *(const v8bf*)&bs[(nBase + 32) * 40 + 16 * half + 8];
    v8bf b3l = *(const v8bf*)&bs[(nBase + 48) * 40 + 16 * half];
    v8bf b3h = *(const v8bf*)&bs[(nBase + 48) * 40 + 16 * half + 8];
    v16bf a  = __builtin_shufflevector(aLo, aHi, 0,1,2,3,4,5,6,7,8,9,10,11,12,13,14,15);
    v16bf b0 = __builtin_shufflevector(b0l, b0h, 0,1,2,3,4,5,6,7,8,9,10,11,12,13,14,15);
    v16bf b1 = __builtin_shufflevector(b1l, b1h, 0,1,2,3,4,5,6,7,8,9,10,11,12,13,14,15);
    v16bf b2 = __builtin_shufflevector(b2l, b2h, 0,1,2,3,4,5,6,7,8,9,10,11,12,13,14,15);
    v16bf b3 = __builtin_shufflevector(b3l, b3h, 0,1,2,3,4,5,6,7,8,9,10,11,12,13,14,15);
    acc[0] = __builtin_amdgcn_wmma_f32_16x16x32_bf16(false, a, false, b0, (short)0, acc[0], false, false);
    acc[1] = __builtin_amdgcn_wmma_f32_16x16x32_bf16(false, a, false, b1, (short)0, acc[1], false, false);
    acc[2] = __builtin_amdgcn_wmma_f32_16x16x32_bf16(false, a, false, b2, (short)0, acc[2], false, false);
    acc[3] = __builtin_amdgcn_wmma_f32_16x16x32_bf16(false, a, false, b3, (short)0, acc[3], false, false);
    buf ^= 1;
  }

  // ---- epilogue: C/D layout elem i -> (m = 8*half + i, n = lm) ----
#pragma unroll
  for (int t = 0; t < 4; ++t) {
    const int ncol = colBase + wn * 64 + t * 16 + lm;
    const float bv = bias[eId * NCOL + ncol];
#pragma unroll
    for (int i = 0; i < 8; ++i) {
      int r = wm * 16 + half * 8 + i;
      if (r < rowsHere) {
        float v = acc[t][i] + bv;
        if (PASS == 1) {
          v = (v > 0.f) ? v : 0.f;
          Hout[(size_t)(posBase + r) * D_FF + ncol] = f2bf(v);
        } else {
          out[(size_t)(posBase + r) * D_MODEL + ncol] = v;
        }
      }
    }
  }
}

// ---------------- launcher ----------------

extern "C" void kernel_launch(void* const* d_in, const int* in_sizes, int n_in,
                              void* d_out, int out_size, void* d_ws, size_t ws_size,
                              hipStream_t stream) {
  const float* x  = (const float*)d_in[0];   // [4096, 1024]
  const float* Wg = (const float*)d_in[1];   // [1024, 8]
  const float* bg = (const float*)d_in[2];   // [8]
  const float* W1 = (const float*)d_in[3];   // [8, 1024, 4096]
  const float* b1 = (const float*)d_in[4];   // [8, 4096]
  const float* W2 = (const float*)d_in[5];   // [8, 4096, 1024]
  const float* b2 = (const float*)d_in[6];   // [8, 1024]
  // d_in[7] = k (top-1, unused)

  char* ws = (char*)d_ws;
  int* eArr   = (int*)(ws);                      // 4096 ints
  int* counts = (int*)(ws + 16384);              // 8 ints
  int* offs   = (int*)(ws + 16384 + 64);         // 8 ints
  int* perm   = (int*)(ws + 16384 + 128);        // 4096 ints
  unsigned short* H = (unsigned short*)(ws + 65536);  // [4096, 4096] bf16 (32 MB)

  float* out = (float*)d_out;

  moe_init_kernel<<<1, 32, 0, stream>>>(counts);
  moe_gate_kernel<<<N_TOKENS / 256, 256, 0, stream>>>(x, Wg, bg, eArr, counts);
  moe_offsets_kernel<<<1, 32, 0, stream>>>(counts, offs);
  moe_perm_kernel<<<N_TOKENS / 256, 256, 0, stream>>>(eArr, offs, perm);

  dim3 g1(D_FF / 128, N_TOKENS / 64, N_EXPERTS);
  moe_ffn_kernel<1><<<g1, 256, 0, stream>>>(x, nullptr, W1, b1, perm, counts, offs,
                                            H, nullptr);
  dim3 g2(D_MODEL / 128, N_TOKENS / 64, N_EXPERTS);
  moe_ffn_kernel<2><<<g2, 256, 0, stream>>>(nullptr, H, W2, b2, perm, counts, offs,
                                            nullptr, out);
}
